// MultiStageModel_53317724013317
// MI455X (gfx1250) — matrix operand order repaired and checked by reading, not compiled, not run on hardware
//
#include <hip/hip_runtime.h>
#include <hip/hip_bf16.h>

// MI455X / gfx1250 implementation.
// Layout convention: every activation tensor stays (B, C, T) with row stride T,
// so every einsum/1x1-conv in the reference is a channel GEMM over contiguous
// time columns -> mapped onto v_wmma_f32_16x16x32_f16 tiles (M=co, N=t, K=ci).
// GEMM X-tiles are staged into LDS by the Tensor Data Mover (tensor_load_to_lds,
// double-buffered, synchronized with s_wait_tensorcnt). The selective scan is a
// 3-phase chunked parallel scan (affine recurrence).

typedef __attribute__((ext_vector_type(16))) _Float16     v16h;
typedef __attribute__((ext_vector_type(8)))  float        v8f;
typedef __attribute__((ext_vector_type(4)))  unsigned int v4u;
typedef __attribute__((ext_vector_type(8)))  int          v8i;
typedef __attribute__((ext_vector_type(4)))  int          v4i;

#define T_LEN   8192
#define NBATCH  2
#define NCHUNK  32
#define CHUNK_L (T_LEN / NCHUNK)

// ---- Tensor Data Mover availability / toolchain arity dispatch -------------
#if defined(__AMDGCN__) && __has_builtin(__builtin_amdgcn_tensor_load_to_lds) && \
    __has_builtin(__builtin_amdgcn_s_wait_tensorcnt)
  #define USE_TDM 1
  #if __has_include(<hip/amd_detail/amd_gfx1250_TDM.h>)
    #define CDNA5_TDM_6ARG 1   // therock-10.0 headers -> clang-23 6-arg builtin
  #else
    #define CDNA5_TDM_6ARG 0   // ROCm 7.2 clang-22 5-arg builtin
  #endif
#else
  #define USE_TDM 0
#endif

__device__ __forceinline__ float act_apply(float x, int act) {
  if (act == 1) return x / (1.0f + __expf(-x));                    // silu
  if (act == 2) return (x > 20.0f) ? x : log1pf(__expf(x));        // softplus
  return x;
}

#if USE_TDM
// Issue one TDM 2D tile load: rows_valid x 64 f32 tile, row stride T_LEN,
// hardware zero-fill for rows >= rows_valid (tensor_dim1 OOB -> 0).
// D# bitfields per CDNA5 ISA 8.3/8.4 (group0: count/lds/global/type,
// group1: data_size, dims, tile dims, dim0 stride). Groups 2/3 zero (2D tile).
__device__ __forceinline__ void tdm_load_tile(const float* gsrc,
                                              unsigned lds_byte_addr,
                                              int rows_valid)
{
  unsigned long long ga = (unsigned long long)(size_t)gsrc;
  v4u g0;
  g0[0] = 1u;                                          // count=1 (valid), user mode
  g0[1] = lds_byte_addr;                               // lds_addr (bytes)
  g0[2] = (unsigned)(ga & 0xffffffffu);                // global_addr[31:0]
  g0[3] = (unsigned)((ga >> 32) & 0x01ffffffu)         // global_addr[56:32]
        | (2u << 30);                                  // type = 2 ("image")
  const unsigned td0 = 64u;                            // tensor_dim0 (>= tile width)
  const unsigned td1 = (unsigned)rows_valid;           // tensor_dim1 -> K-tail zero-fill
  v8i g1;
  g1[0] = (int)(2u << 16);                             // wg_mask=0, data_size=2 (4B)
  g1[1] = (int)((td0 & 0xffffu) << 16);                // atomic_bar=0 | tensor_dim0 lo16
  g1[2] = (int)(((td0 >> 16) & 0xffffu) | ((td1 & 0xffffu) << 16)); // td0 hi | td1 lo
  g1[3] = (int)(((td1 >> 16) & 0xffffu) | (64u << 16));             // td1 hi | tile_dim0=64
  g1[4] = (int)(32u);                                  // tile_dim1=32, tile_dim2=0
  g1[5] = (int)(T_LEN);                                // tensor_dim0_stride[31:0]
  g1[6] = 0;                                           // stride0 hi16 | stride1 lo16
  g1[7] = 0;                                           // stride1 hi32
  v4i z4 = {0, 0, 0, 0};
#if CDNA5_TDM_6ARG
  v8i z8 = {0, 0, 0, 0, 0, 0, 0, 0};
  __builtin_amdgcn_tensor_load_to_lds(g0, g1, z4, z4, z8, 0);
#else
  __builtin_amdgcn_tensor_load_to_lds(g0, g1, z4, z4, 0);
#endif
}
#endif  // USE_TDM

// ---------------------------------------------------------------------------
// Generic pointwise-conv / channel GEMM:
//   Y[b,co,t] = act(bias[co] + sum_ci W[co,ci] * X[b,ci,t])
// Block = 128 threads = 4 waves. Each wave owns a 16(co) x 64(t) strip as four
// 16x16 WMMA tiles; K (ci) walked in chunks of 32. X slabs are staged in LDS by
// the TDM (double-buffered, wave 0 issues, TENSORcnt + barrier synchronize).
// ---------------------------------------------------------------------------
__global__ __launch_bounds__(128)
void gemm_pw_wmma(const float* __restrict__ X, long xbs, int Ci,
                  const float* __restrict__ W,      // (Co, Ci) row-major
                  const float* __restrict__ bias,   // (Co) or nullptr
                  float* __restrict__ Y, long ybs, int Co,
                  int act)
{
  __shared__ float lds[2][32 * 64];  // two K-chunk x t-tile buffers, 16 KB

  const int b    = blockIdx.z;
  const int t0   = blockIdx.x * 64;
  const int cb   = blockIdx.y * 64;
  const int tid  = threadIdx.x;
  const int wave = tid >> 5;
  const int lane = tid & 31;
  const int n16  = lane & 15;
  const bool hi  = lane >= 16;
  const int co0  = cb + wave * 16;
  const int arow = co0 + n16;            // A-matrix row this lane feeds
  const int kSub = hi ? 8 : 0;           // ISA 16-bit A layout K sub-offset

  const float* Xb = X + (size_t)b * xbs + t0;   // row ci at Xb + ci*T_LEN

  v8f acc[4];
  #pragma unroll
  for (int j = 0; j < 4; ++j)
    #pragma unroll
    for (int v = 0; v < 8; ++v) acc[j][v] = 0.0f;

#if USE_TDM
  const unsigned lds0 = (unsigned)(size_t)(&lds[0][0]);  // LDS byte offset, buf 0
  if (wave == 0)
    tdm_load_tile(Xb, lds0, (Ci < 32) ? Ci : 32);
#endif

  int cur = 0;
  for (int k0 = 0; k0 < Ci; k0 += 32, cur ^= 1) {
#if USE_TDM
    if (wave == 0) {
      const int k1 = k0 + 32;
      if (k1 < Ci) {   // kick next slab into the other buffer, keep 1 in flight
        int rem = Ci - k1;
        tdm_load_tile(Xb + (size_t)k1 * T_LEN,
                      lds0 + (unsigned)((cur ^ 1) * 32 * 64 * 4),
                      (rem < 32) ? rem : 32);
        __builtin_amdgcn_s_wait_tensorcnt((short)1);
      } else {
        __builtin_amdgcn_s_wait_tensorcnt((short)0);
      }
    }
    __syncthreads();   // publish TDM-written slab to all waves
#else
    // ---- synchronous fallback staging (host pass / no-TDM toolchains) ----
    #pragma unroll
    for (int i = 0; i < 16; ++i) {
      int idx = tid + i * 128;
      int row = idx >> 6, col = idx & 63;
      int ci  = k0 + row;
      lds[cur][idx] = (ci < Ci) ? Xb[(size_t)ci * T_LEN + col] : 0.0f;
    }
    __syncthreads();
#endif

    // ---- A fragment from W (16-bit A 16x32 layout, ISA 7.12.2) ----------
    v16h a;
    #pragma unroll
    for (int j = 0; j < 8; ++j) {
      int k1 = k0 + kSub + j;
      int k2 = k1 + 16;
      float w1 = (arow < Co && k1 < Ci) ? W[(size_t)arow * Ci + k1] : 0.0f;
      float w2 = (arow < Co && k2 < Ci) ? W[(size_t)arow * Ci + k2] : 0.0f;
      a[j]     = (_Float16)w1;
      a[8 + j] = (_Float16)w2;
    }

    // ---- B fragments from LDS (K rows kSub+0..7 | 16+kSub+0..7 per half-wave) -----
    const float* slab = &lds[cur][0];
    #pragma unroll
    for (int j = 0; j < 4; ++j) {
      v16h bf;
      const int col = j * 16 + n16;
      #pragma unroll
      for (int i = 0; i < 16; ++i)
        bf[i] = (_Float16)slab[((i < 8) ? (kSub + i) : (8 + kSub + i)) * 64 + col];
      acc[j] = __builtin_amdgcn_wmma_f32_16x16x32_f16(
                   false, a, false, bf, (short)0, acc[j], false, false);
    }
    __syncthreads();   // all waves done with this slab before it is reused
  }

  // ---- epilogue: bias + activation + store (D layout: m = v + hi*8) -----
  const int mBase = hi ? 8 : 0;
  float* Yb = Y + (size_t)b * ybs;
  #pragma unroll
  for (int v = 0; v < 8; ++v) {
    int m = co0 + mBase + v;
    if (m < Co) {
      float bv = bias ? bias[m] : 0.0f;
      #pragma unroll
      for (int j = 0; j < 4; ++j) {
        float r = acc[j][v] + bv;
        Yb[(size_t)m * T_LEN + t0 + j * 16 + n16] = act_apply(r, act);
      }
    }
  }
}

// ---------------------------------------------------------------------------
// Small dilated conv, kernel width 3, "same" padding (pad = dil).
// ---------------------------------------------------------------------------
__global__ void dilconv3(const float* __restrict__ X, long xbs, int Ci,
                         const float* __restrict__ W,   // (Co, Ci, 3)
                         const float* __restrict__ bias,
                         float* __restrict__ Y, long ybs, int dil)
{
  const int t  = blockIdx.x * blockDim.x + threadIdx.x;
  const int co = blockIdx.y;
  const int b  = blockIdx.z;
  const float* Xb = X + (size_t)b * xbs;
  float acc = bias[co];
  for (int ci = 0; ci < Ci; ++ci) {
    const float* xr = Xb + (size_t)ci * T_LEN;
    const float* wr = W + ((size_t)co * Ci + ci) * 3;
    #pragma unroll
    for (int k = 0; k < 3; ++k) {
      int tt = t + dil * (k - 1);
      if (tt >= 0 && tt < T_LEN) acc += wr[k] * xr[tt];
    }
  }
  Y[(size_t)b * ybs + (size_t)co * T_LEN + t] = acc;
}

// ---------------------------------------------------------------------------
// Depthwise causal conv (width 4, left pad 3) + SiLU.  xi = channels 0..127 of xz.
// ---------------------------------------------------------------------------
__global__ void dwconv_silu(const float* __restrict__ XZ,   // (B,256,T)
                            const float* __restrict__ W4,   // (128,1,4)
                            const float* __restrict__ bias, // (128)
                            float* __restrict__ XI2)        // (B,128,T)
{
  const int t = blockIdx.x * blockDim.x + threadIdx.x;
  const int d = blockIdx.y;
  const int b = blockIdx.z;
  const float* xr = XZ + ((size_t)b * 256 + d) * T_LEN;
  float acc = bias[d];
  #pragma unroll
  for (int k = 0; k < 4; ++k) {
    int tt = t + k - 3;
    if (tt >= 0) acc += W4[d * 4 + k] * xr[tt];
  }
  XI2[((size_t)b * 128 + d) * T_LEN + t] = acc / (1.0f + __expf(-acc));
}

// ---------------------------------------------------------------------------
// Selective scan, chunked parallel form. h_t = exp(dt*A)*h + dt*B_t*u_t.
// Phase 1: per (b,d,n,chunk) compute chunk aggregate (P = prod dA, S = state from h0=0).
// ---------------------------------------------------------------------------
__global__ void scan_phase1(const float* __restrict__ delta,  // (B,128,T)
                            const float* __restrict__ u,      // (B,128,T)
                            const float* __restrict__ dbc,    // (B,36,T), B at ch 4
                            const float* __restrict__ A_log,  // (128,16)
                            float* __restrict__ P, float* __restrict__ S)
{
  const int n = threadIdx.x;
  const int c = blockIdx.x * blockDim.y + threadIdx.y;
  const int d = blockIdx.y;
  const int b = blockIdx.z;
  const float A = -__expf(A_log[d * 16 + n]);
  const float* dr = delta + ((size_t)b * 128 + d)     * T_LEN + c * CHUNK_L;
  const float* ur = u     + ((size_t)b * 128 + d)     * T_LEN + c * CHUNK_L;
  const float* Br = dbc   + ((size_t)b * 36 + 4 + n)  * T_LEN + c * CHUNK_L;
  float h = 0.0f, p = 1.0f;
  for (int t = 0; t < CHUNK_L; ++t) {
    float dl = dr[t];
    float dA = __expf(dl * A);
    h = dA * h + dl * Br[t] * ur[t];
    p *= dA;
  }
  size_t idx = (((size_t)b * 128 + d) * 16 + n) * NCHUNK + c;
  P[idx] = p; S[idx] = h;
}

// Phase 2: tiny sequential pass over chunks -> per-chunk entry state.
__global__ void scan_phase2(const float* __restrict__ P, const float* __restrict__ S,
                            float* __restrict__ Hin)
{
  const int i = blockIdx.x * blockDim.x + threadIdx.x;   // (b*128+d)*16+n
  const size_t base = (size_t)i * NCHUNK;
  float h = 0.0f;
  for (int c = 0; c < NCHUNK; ++c) {
    Hin[base + c] = h;
    h = P[base + c] * h + S[base + c];
  }
}

// Phase 3: replay each chunk from its entry state; 16 lanes = 16 states,
// cross-state C-reduction via shfl_xor; fuse +u*D and the silu(z) gate.
__global__ __launch_bounds__(32)
void scan_phase3(const float* __restrict__ delta,
                 const float* __restrict__ u,
                 const float* __restrict__ dbc,    // B at ch 4, C at ch 20
                 const float* __restrict__ A_log,
                 const float* __restrict__ Dv,     // (128)
                 const float* __restrict__ XZ,     // z at ch 128+d
                 const float* __restrict__ Hin,
                 float* __restrict__ Yout)         // (B,128,T)
{
  const int lane = threadIdx.x;
  const int n = lane & 15;
  const int c = blockIdx.x * 2 + (lane >> 4);
  const int d = blockIdx.y;
  const int b = blockIdx.z;
  const float A  = -__expf(A_log[d * 16 + n]);
  const float Dd = Dv[d];
  const float* dr = delta + ((size_t)b * 128 + d)      * T_LEN + c * CHUNK_L;
  const float* ur = u     + ((size_t)b * 128 + d)      * T_LEN + c * CHUNK_L;
  const float* Br = dbc   + ((size_t)b * 36 + 4 + n)   * T_LEN + c * CHUNK_L;
  const float* Cr = dbc   + ((size_t)b * 36 + 20 + n)  * T_LEN + c * CHUNK_L;
  const float* zr = XZ    + ((size_t)b * 256 + 128 + d)* T_LEN + c * CHUNK_L;
  float*       yr = Yout  + ((size_t)b * 128 + d)      * T_LEN + c * CHUNK_L;
  float h = Hin[(((size_t)b * 128 + d) * 16 + n) * NCHUNK + c];
  for (int t = 0; t < CHUNK_L; ++t) {
    float dl = dr[t];
    float ut = ur[t];
    float dA = __expf(dl * A);
    h = dA * h + dl * Br[t] * ut;
    float val = h * Cr[t];
    #pragma unroll
    for (int m = 1; m < 16; m <<= 1)        // stays inside 16-lane n-group
      val += __shfl_xor(val, m, 32);
    if (n == 0) {
      float z  = zr[t];
      float sz = z / (1.0f + __expf(-z));
      yr[t] = (val + ut * Dd) * sz;
    }
  }
}

// ---------------------------------------------------------------------------
// Final 1x96 classifier head.
// ---------------------------------------------------------------------------
__global__ void cls_head(const float* __restrict__ X,   // (B,96,T)
                         const float* __restrict__ Wc,  // (96)
                         const float* __restrict__ bc,  // (1)
                         float* __restrict__ out)       // (B,1,T)
{
  const int t = blockIdx.x * blockDim.x + threadIdx.x;
  const int b = blockIdx.y;
  const float* Xb = X + (size_t)b * 96 * T_LEN;
  float acc = bc[0];
  for (int ci = 0; ci < 96; ++ci) acc += Wc[ci] * Xb[(size_t)ci * T_LEN + t];
  out[(size_t)b * T_LEN + t] = acc;
}

// ---------------------------------------------------------------------------
extern "C" void kernel_launch(void* const* d_in, const int* in_sizes, int n_in,
                              void* d_out, int out_size, void* d_ws, size_t ws_size,
                              hipStream_t stream)
{
  (void)in_sizes; (void)n_in; (void)out_size; (void)ws_size;
  const float* x = (const float*)d_in[0];

  // Bump allocator over d_ws (total ~98 MB).
  char* ws = (char*)d_ws;
  size_t off = 0;
  auto alloc = [&](size_t bytes) -> float* {
    size_t r = off;
    off = (off + bytes + 255) & ~(size_t)255;
    return (float*)(ws + r);
  };
  float* io0  = alloc((size_t)NBATCH * 384 * T_LEN * 4);  // stage0 out / stage2 out
  float* io1  = alloc((size_t)NBATCH * 192 * T_LEN * 4);  // stage1 out
  float* cat  = alloc((size_t)NBATCH * 88  * T_LEN * 4);  // [out|x1|x2|x3]
  float* mo   = alloc((size_t)NBATCH * 64  * T_LEN * 4);  // mix out = mamba in
  float* xz   = alloc((size_t)NBATCH * 256 * T_LEN * 4);  // in_proj out (xi|z)
  float* xi2  = alloc((size_t)NBATCH * 128 * T_LEN * 4);  // silu(depthwise conv)
  float* dbc  = alloc((size_t)NBATCH * 36  * T_LEN * 4);  // [dt|B|C]
  float* dlt  = alloc((size_t)NBATCH * 128 * T_LEN * 4);  // softplus(dt_proj)
  float* ybuf = alloc((size_t)NBATCH * 128 * T_LEN * 4);  // gated scan output
  float* mout = alloc((size_t)NBATCH * 64  * T_LEN * 4);  // out_proj output
  float* Pb   = alloc((size_t)NBATCH * 128 * 16 * NCHUNK * 4);
  float* Sb   = alloc((size_t)NBATCH * 128 * 16 * NCHUNK * 4);
  float* Hb   = alloc((size_t)NBATCH * 128 * 16 * NCHUNK * 4);

  const float* sin = x;
  int din = 768;
  const dim3 gT64(T_LEN / 64, 1, NBATCH);

  for (int s = 0; s < 3; ++s) {
    const float* const* p = (const float* const*)(d_in + 1 + 21 * s);
    // p[0..11]: w_in,b_in,w1,b1,w2,b2,w3,b3,w_mix,b_mix,w_out,b_out
    // p[12..20]: in_proj_W,conv1d_W,conv1d_b,x_proj_W,dt_proj_W,dt_proj_b,A_log,D,out_proj_W
    const int dout = din >> 1;
    float* sout = (s == 0) ? io0 : (s == 1) ? io1 : io0;

    // w_in (1x1) -> cat channels 0..63
    gemm_pw_wmma<<<gT64, 128, 0, stream>>>(sin, (long)din * T_LEN, din,
                                           p[0], p[1], cat, 88L * T_LEN, 64, 0);
    // dilated convs -> cat channels 64..87
    dilconv3<<<dim3(T_LEN / 256, 8, NBATCH), 256, 0, stream>>>(
        cat,              88L * T_LEN, 64, p[2], p[3], cat + 64 * T_LEN, 88L * T_LEN, 2);
    dilconv3<<<dim3(T_LEN / 256, 8, NBATCH), 256, 0, stream>>>(
        cat + 64 * T_LEN, 88L * T_LEN, 8,  p[4], p[5], cat + 72 * T_LEN, 88L * T_LEN, 4);
    dilconv3<<<dim3(T_LEN / 256, 8, NBATCH), 256, 0, stream>>>(
        cat + 72 * T_LEN, 88L * T_LEN, 8,  p[6], p[7], cat + 80 * T_LEN, 88L * T_LEN, 8);
    // w_mix (1x1 over 88 ch)
    gemm_pw_wmma<<<gT64, 128, 0, stream>>>(cat, 88L * T_LEN, 88,
                                           p[8], p[9], mo, 64L * T_LEN, 64, 0);
    // mamba in_proj: (256,64)
    gemm_pw_wmma<<<dim3(T_LEN / 64, 4, NBATCH), 128, 0, stream>>>(
        mo, 64L * T_LEN, 64, p[12], nullptr, xz, 256L * T_LEN, 256, 0);
    // depthwise causal conv + silu
    dwconv_silu<<<dim3(T_LEN / 256, 128, NBATCH), 256, 0, stream>>>(xz, p[13], p[14], xi2);
    // x_proj: (36,128) -> [dt|B|C]
    gemm_pw_wmma<<<gT64, 128, 0, stream>>>(xi2, 128L * T_LEN, 128,
                                           p[15], nullptr, dbc, 36L * T_LEN, 36, 0);
    // dt_proj + softplus: (128,4) on dbc ch 0..3
    gemm_pw_wmma<<<dim3(T_LEN / 64, 2, NBATCH), 128, 0, stream>>>(
        dbc, 36L * T_LEN, 4, p[16], p[17], dlt, 128L * T_LEN, 128, 2);
    // chunked selective scan (+ fused skip and silu(z) gate)
    scan_phase1<<<dim3(NCHUNK / 8, 128, NBATCH), dim3(16, 8), 0, stream>>>(
        dlt, xi2, dbc, p[18], Pb, Sb);
    scan_phase2<<<dim3(16), 256, 0, stream>>>(Pb, Sb, Hb);
    scan_phase3<<<dim3(NCHUNK / 2, 128, NBATCH), 32, 0, stream>>>(
        dlt, xi2, dbc, p[18], p[19], xz, Hb, ybuf);
    // out_proj: (64,128)
    gemm_pw_wmma<<<gT64, 128, 0, stream>>>(ybuf, 128L * T_LEN, 128,
                                           p[20], nullptr, mout, 64L * T_LEN, 64, 0);
    // w_out (1x1): (dout,64)
    gemm_pw_wmma<<<dim3(T_LEN / 64, (dout + 63) / 64, NBATCH), 128, 0, stream>>>(
        mout, 64L * T_LEN, 64, p[10], p[11], sout, (long)dout * T_LEN, dout, 0);

    sin = sout; din = dout;
  }

  cls_head<<<dim3(T_LEN / 256, NBATCH), 256, 0, stream>>>(
      sin, (const float*)d_in[64], (const float*)d_in[65], (float*)d_out);
}
